// DLCC_52845277610207
// MI455X (gfx1250) — matrix operand order repaired and checked by reading, not compiled, not run on hardware
//
#include <hip/hip_runtime.h>

// Problem constants (from reference setup_inputs)
#define B_    8
#define C_    384
#define HW_   56
#define N_    (HW_ * HW_)     // 3136
#define NH_   8
#define HD_   48
#define NS_   9
#define MT_   (B_ * N_)       // 25088 rows
#define K3C_  (3 * C_)        // 1152

typedef __attribute__((ext_vector_type(16))) __bf16 v16bf;
typedef __attribute__((ext_vector_type(8)))  __bf16 v8bf;
typedef __attribute__((ext_vector_type(8)))  float  v8f;

// fp32 -> bf16 (round-to-nearest-even) via bit ops; storage is ushort so we
// never rely on __bf16 scalar arithmetic, only on the vector WMMA operand type.
__device__ __forceinline__ unsigned short f2bf(float f) {
  unsigned int u = __float_as_uint(f);
  unsigned int r = 0x7fffu + ((u >> 16) & 1u);
  u += r;
  return (unsigned short)(u >> 16);
}

__global__ void dlcc_cvt_bf16(const float* __restrict__ in,
                              unsigned short* __restrict__ out, int n) {
  int i = blockIdx.x * blockDim.x + threadIdx.x;
  if (i < n) out[i] = f2bf(in[i]);
}

// x (B,C,N) fp32 -> xf (B,N,C) bf16, 32x32 LDS tile transpose
__global__ void dlcc_transpose_x(const float* __restrict__ x,
                                 unsigned short* __restrict__ xf) {
  __shared__ float tile[32][33];
  const int n0 = blockIdx.x * 32;
  const int c0 = blockIdx.y * 32;
  const int b  = blockIdx.z;
  const int tx = threadIdx.x;   // 0..31
  const int ty = threadIdx.y;   // 0..7
#pragma unroll
  for (int r = ty; r < 32; r += 8)
    tile[r][tx] = x[((size_t)b * C_ + c0 + r) * N_ + n0 + tx];
  __syncthreads();
#pragma unroll
  for (int r = ty; r < 32; r += 8)
    xf[((size_t)b * N_ + n0 + r) * C_ + c0 + tx] = f2bf(tile[tx][r]);
}

// ---------------------------------------------------------------------------
// WMMA bf16 GEMM #1:  Out(MT_,1152) = A(MT_,384) * Wt(1152,384)^T
// Wave tile: 16(M) x 32(N), 2 f32 accumulators, K stepped by 32.
// Fragment layouts (A and B both use contiguous k = 16*half + s, not the ISA table order):
//   A 16x32 bf16 : lane l%16 = row M; lanes<16 hold K kb..kb+15,
//                  lanes>=16 hold K kb+16..kb+31  -> two contiguous 16B loads.
//   B 32x16 bf16 : lane l%16 = col N; lane half selects K block of 16
//                  -> one contiguous 32B load from Wt[n, kb+16*half .. +16).
//   C/D 16x16 f32: vgpr r -> M = r + 8*(l/16), N = l%16.
// ---------------------------------------------------------------------------
__global__ __launch_bounds__(256) void dlcc_gemm_qkv(
    const unsigned short* __restrict__ A,    // (MT_, C_) bf16
    const unsigned short* __restrict__ Wt,   // (K3C_, C_) bf16 row-major
    float* __restrict__ Out)                 // (MT_, K3C_) f32
{
  const int lane  = threadIdx.x & 31;
  const int wave  = threadIdx.x >> 5;        // 0..7
  const int lmod  = lane & 15;
  const int lhalf = lane >> 4;
  const int m0 = blockIdx.x * 32 + (wave >> 2) * 16;
  const int n0 = blockIdx.y * 128 + (wave & 3) * 32;

  const unsigned short* arow = A  + (size_t)(m0 + lmod) * C_ + 16 * lhalf;
  const unsigned short* b0p  = Wt + (size_t)(n0 + lmod) * C_ + 16 * lhalf;
  const unsigned short* b1p  = Wt + (size_t)(n0 + 16 + lmod) * C_ + 16 * lhalf;

  v8f acc0 = {};
  v8f acc1 = {};
#pragma unroll
  for (int kb = 0; kb < C_; kb += 32) {
    if (kb + 128 < C_) {   // folds at compile time (loop fully unrolled)
      __builtin_prefetch(arow + kb + 128, 0, 0);
      __builtin_prefetch(b0p  + kb + 128, 0, 0);
      __builtin_prefetch(b1p  + kb + 128, 0, 0);
    }
    v8bf alo = *(const v8bf*)(arow + kb);
    v8bf ahi = *(const v8bf*)(arow + kb + 8);
    v16bf af = __builtin_shufflevector(alo, ahi,
        0,1,2,3,4,5,6,7,8,9,10,11,12,13,14,15);
    v16bf bf0 = *(const v16bf*)(b0p + kb);
    v16bf bf1 = *(const v16bf*)(b1p + kb);
    acc0 = __builtin_amdgcn_wmma_f32_16x16x32_bf16(
        false, af, false, bf0, (short)0, acc0, false, false);
    acc1 = __builtin_amdgcn_wmma_f32_16x16x32_bf16(
        false, af, false, bf1, (short)0, acc1, false, false);
  }

  float* orow = Out + (size_t)(m0 + 8 * lhalf) * K3C_ + n0 + lmod;
#pragma unroll
  for (int r = 0; r < 8; ++r) {
    orow[(size_t)r * K3C_]      = acc0[r];
    orow[(size_t)r * K3C_ + 16] = acc1[r];
  }
}

// Per-head sampling-offset projection: offs[(b*nh+h)*N+n][j] ,
// j in [0,18): (dy,dx) pairs per tap.  18*48 MACs per thread.
__global__ void dlcc_offsets(const float* __restrict__ x,
                             const float* __restrict__ Woff,   // (18, 48)
                             float* __restrict__ offs) {
  int idx = blockIdx.x * blockDim.x + threadIdx.x;
  const int total = B_ * NH_ * N_;
  if (idx >= total) return;
  int n  = idx % N_;
  int bh = idx / N_;
  int h  = bh % NH_;
  int b  = bh / NH_;
  const float* xp = x + ((size_t)b * C_ + h * HD_) * N_ + n;
  float xv[HD_];
#pragma unroll
  for (int d = 0; d < HD_; ++d) xv[d] = xp[(size_t)d * N_];
  float* op = offs + (size_t)idx * (2 * NS_);
  for (int j = 0; j < 2 * NS_; ++j) {
    const float* w = Woff + j * HD_;
    float s = 0.f;
#pragma unroll
    for (int d = 0; d < HD_; ++d) s += xv[d] * w[d];
    op[j] = s;
  }
}

// ---------------------------------------------------------------------------
// Deformable attention, pass 1: per (b,h,n) compute the 9 tap logits
// (bilinear-weighted q.k dots; bilinear commutes with the dot product),
// softmax, and store NORMALIZED probabilities.  Only q[48] lives in
// registers -> much better occupancy for this gather-latency-bound pass.
// qkv row layout: [q(h*48) | k(384+h*48) | v(768+h*48)] within 1152 floats.
// ---------------------------------------------------------------------------
__global__ void dlcc_attn_logits(const float* __restrict__ qkv,
                                 const float* __restrict__ offs,
                                 float* __restrict__ probs) {   // (B*nh*N, 9)
  int idx = blockIdx.x * blockDim.x + threadIdx.x;
  const int total = B_ * NH_ * N_;
  if (idx >= total) return;
  int n  = idx % N_;
  int bh = idx / N_;
  int h  = bh % NH_;
  int b  = bh / NH_;
  const size_t mbase = (size_t)b * N_;
  const float* qp = qkv + (mbase + n) * K3C_ + h * HD_;
  float q[HD_];
#pragma unroll
  for (int d = 0; d < HD_; ++d) q[d] = qp[d];

  const int iy = n / HW_;
  const int ix = n % HW_;
  const float* op = offs + (size_t)idx * (2 * NS_);
  const int koff = (NH_ + h) * HD_;
  const float scale = 0.144337567297406f;   // 48^-0.5

  float logit[NS_];
  for (int s = 0; s < NS_; ++s) {
    float ys = (float)(iy + s / 3 - 1) + op[2 * s];
    float xs = (float)(ix + s % 3 - 1) + op[2 * s + 1];
    float y0f = floorf(ys), x0f = floorf(xs);
    float wy = ys - y0f, wx = xs - x0f;
    int y0 = (int)y0f, x0 = (int)x0f;
    float acc = 0.f;
    for (int cy = 0; cy < 2; ++cy) {
      int yy = y0 + cy;
      if (yy < 0 || yy >= HW_) continue;
      float wyc = cy ? wy : 1.f - wy;
      for (int cx = 0; cx < 2; ++cx) {
        int xx = x0 + cx;
        if (xx < 0 || xx >= HW_) continue;
        float wgt = wyc * (cx ? wx : 1.f - wx);
        const float* kp = qkv + (mbase + yy * HW_ + xx) * K3C_ + koff;
        float dot = 0.f;
#pragma unroll
        for (int d = 0; d < HD_; ++d) dot += q[d] * kp[d];
        acc += wgt * dot;
      }
    }
    logit[s] = acc * scale;
  }

  float mx = logit[0];
#pragma unroll
  for (int s = 1; s < NS_; ++s) mx = fmaxf(mx, logit[s]);
  float p[NS_];
  float denom = 0.f;
#pragma unroll
  for (int s = 0; s < NS_; ++s) { p[s] = __expf(logit[s] - mx); denom += p[s]; }
  float inv = 1.f / denom;
  float* pp = probs + (size_t)idx * NS_;
#pragma unroll
  for (int s = 0; s < NS_; ++s) pp[s] = p[s] * inv;
}

// ---------------------------------------------------------------------------
// Deformable attention, pass 2: one thread per (b,h,n, 16-channel chunk).
// out[16] + p[9] only -> tiny register footprint, high occupancy to hide the
// bilinear V-gather latency.  Writes the (MT_, C_) bf16 A-operand for proj.
// ---------------------------------------------------------------------------
__global__ void dlcc_attn_out(const float* __restrict__ qkv,
                              const float* __restrict__ offs,
                              const float* __restrict__ probs,
                              unsigned short* __restrict__ aout) { // (MT_, C_)
  int t = blockIdx.x * blockDim.x + threadIdx.x;
  const int total = B_ * NH_ * N_ * 3;
  if (t >= total) return;
  const int chunk = t % 3;          // 16-channel chunk within the head
  const int idx   = t / 3;          // (b,h,n)
  int n  = idx % N_;
  int bh = idx / N_;
  int h  = bh % NH_;
  int b  = bh / NH_;
  const size_t mbase = (size_t)b * N_;

  const int iy = n / HW_;
  const int ix = n % HW_;
  const float* op = offs + (size_t)idx * (2 * NS_);
  const float* pp = probs + (size_t)idx * NS_;
  float p[NS_];
#pragma unroll
  for (int s = 0; s < NS_; ++s) p[s] = pp[s];

  const int voff = (2 * NH_ + h) * HD_ + chunk * 16;
  float out[16];
#pragma unroll
  for (int d = 0; d < 16; ++d) out[d] = 0.f;

  for (int s = 0; s < NS_; ++s) {
    float ys = (float)(iy + s / 3 - 1) + op[2 * s];
    float xs = (float)(ix + s % 3 - 1) + op[2 * s + 1];
    float y0f = floorf(ys), x0f = floorf(xs);
    float wy = ys - y0f, wx = xs - x0f;
    int y0 = (int)y0f, x0 = (int)x0f;
    for (int cy = 0; cy < 2; ++cy) {
      int yy = y0 + cy;
      if (yy < 0 || yy >= HW_) continue;
      float wyc = cy ? wy : 1.f - wy;
      for (int cx = 0; cx < 2; ++cx) {
        int xx = x0 + cx;
        if (xx < 0 || xx >= HW_) continue;
        float wgt = p[s] * wyc * (cx ? wx : 1.f - wx);
        const float* vp = qkv + (mbase + yy * HW_ + xx) * K3C_ + voff;
#pragma unroll
        for (int d = 0; d < 16; ++d) out[d] += wgt * vp[d];
      }
    }
  }
  unsigned short* ap = aout + (mbase + n) * C_ + h * HD_ + chunk * 16;
#pragma unroll
  for (int d = 0; d < 16; ++d) ap[d] = f2bf(out[d]);
}

// WMMA bf16 GEMM #2 + bias, scatter-stored to (B, C, N) output layout.
__global__ __launch_bounds__(256) void dlcc_gemm_proj(
    const unsigned short* __restrict__ A,    // (MT_, C_) bf16
    const unsigned short* __restrict__ Wt,   // (C_, C_) bf16 row-major
    const float* __restrict__ bias,          // (C_)
    float* __restrict__ Out)                 // (B_, C_, N_) f32
{
  const int lane  = threadIdx.x & 31;
  const int wave  = threadIdx.x >> 5;
  const int lmod  = lane & 15;
  const int lhalf = lane >> 4;
  const int m0 = blockIdx.x * 32 + (wave >> 2) * 16;
  const int n0 = blockIdx.y * 128 + (wave & 3) * 32;

  const unsigned short* arow = A  + (size_t)(m0 + lmod) * C_ + 16 * lhalf;
  const unsigned short* b0p  = Wt + (size_t)(n0 + lmod) * C_ + 16 * lhalf;
  const unsigned short* b1p  = Wt + (size_t)(n0 + 16 + lmod) * C_ + 16 * lhalf;

  v8f acc0 = {};
  v8f acc1 = {};
#pragma unroll
  for (int kb = 0; kb < C_; kb += 32) {
    if (kb + 128 < C_) {
      __builtin_prefetch(arow + kb + 128, 0, 0);
      __builtin_prefetch(b0p  + kb + 128, 0, 0);
      __builtin_prefetch(b1p  + kb + 128, 0, 0);
    }
    v8bf alo = *(const v8bf*)(arow + kb);
    v8bf ahi = *(const v8bf*)(arow + kb + 8);
    v16bf af = __builtin_shufflevector(alo, ahi,
        0,1,2,3,4,5,6,7,8,9,10,11,12,13,14,15);
    v16bf bf0 = *(const v16bf*)(b0p + kb);
    v16bf bf1 = *(const v16bf*)(b1p + kb);
    acc0 = __builtin_amdgcn_wmma_f32_16x16x32_bf16(
        false, af, false, bf0, (short)0, acc0, false, false);
    acc1 = __builtin_amdgcn_wmma_f32_16x16x32_bf16(
        false, af, false, bf1, (short)0, acc1, false, false);
  }

  // m = m0 + 8*lhalf + r ; N_ is a multiple of 16 so batch is tile-constant.
  const int m_lo = m0 + 8 * lhalf;
  const int b  = m_lo / N_;
  const int n  = m_lo % N_;
  const int c0 = n0 + lmod;
  const int c1 = n0 + 16 + lmod;
  const float bi0 = bias[c0];
  const float bi1 = bias[c1];
  float* ob = Out + (size_t)b * C_ * N_ + n;
#pragma unroll
  for (int r = 0; r < 8; ++r) {
    ob[(size_t)c0 * N_ + r] = acc0[r] + bi0;
    ob[(size_t)c1 * N_ + r] = acc1[r] + bi1;
  }
}

extern "C" void kernel_launch(void* const* d_in, const int* in_sizes, int n_in,
                              void* d_out, int out_size, void* d_ws, size_t ws_size,
                              hipStream_t stream) {
  const float* x     = (const float*)d_in[0];   // (8,384,56,56)
  const float* Wqkv  = (const float*)d_in[1];   // (1152,384)
  const float* Woff  = (const float*)d_in[2];   // (18,48)
  const float* Wproj = (const float*)d_in[3];   // (384,384)
  const float* bproj = (const float*)d_in[4];   // (384)
  float* out = (float*)d_out;                   // (8,384,56,56)

  char* ws = (char*)d_ws;
  unsigned short* wqkv_bf  = (unsigned short*)(ws);              //    884,736 B
  unsigned short* wproj_bf = (unsigned short*)(ws + 884736);     //    294,912 B
  unsigned short* xf_bf    = (unsigned short*)(ws + 1179648);    // 19,267,584 B
  float*          qkv_f    = (float*)(ws + 20447232);            //115,605,504 B
  float*          offs     = (float*)(ws + 136052736);           // 14,450,688 B
  unsigned short* aout_bf  = (unsigned short*)(ws + 150503424);  // 19,267,584 B
  float*          probs    = (float*)(ws + 169771008);           //  7,225,344 B

  dlcc_cvt_bf16<<<dim3((K3C_ * C_ + 255) / 256), dim3(256), 0, stream>>>(
      Wqkv, wqkv_bf, K3C_ * C_);
  dlcc_cvt_bf16<<<dim3((C_ * C_ + 255) / 256), dim3(256), 0, stream>>>(
      Wproj, wproj_bf, C_ * C_);
  dlcc_transpose_x<<<dim3(N_ / 32, C_ / 32, B_), dim3(32, 8), 0, stream>>>(
      x, xf_bf);
  dlcc_gemm_qkv<<<dim3(MT_ / 32, K3C_ / 128), dim3(256), 0, stream>>>(
      xf_bf, wqkv_bf, qkv_f);
  dlcc_offsets<<<dim3((B_ * NH_ * N_ + 127) / 128), dim3(128), 0, stream>>>(
      x, Woff, offs);
  dlcc_attn_logits<<<dim3((B_ * NH_ * N_ + 127) / 128), dim3(128), 0, stream>>>(
      qkv_f, offs, probs);
  dlcc_attn_out<<<dim3((B_ * NH_ * N_ * 3 + 127) / 128), dim3(128), 0, stream>>>(
      qkv_f, offs, probs, aout_bf);
  dlcc_gemm_proj<<<dim3(MT_ / 32, C_ / 128), dim3(256), 0, stream>>>(
      aout_bf, wproj_bf, bproj, out);
}
